// MultiHeadAttention_35897336660687
// MI455X (gfx1250) — matrix operand order repaired and checked by reading, not compiled, not run on hardware
//
#include <hip/hip_runtime.h>
#include <hip/hip_bf16.h>
#include <cstdint>

typedef __attribute__((ext_vector_type(16))) _Float16 v16h;
typedef __attribute__((ext_vector_type(8)))  _Float16 v8h;
typedef __attribute__((ext_vector_type(4)))  _Float16 v4h;
typedef __attribute__((ext_vector_type(8)))  float    v8f;
typedef __attribute__((ext_vector_type(4)))  unsigned int u32x4;
typedef __attribute__((ext_vector_type(8)))  int      i32x8;
typedef __attribute__((ext_vector_type(4)))  int      i32x4;

#define HIDDEN 1024
#define HEADS  16
#define HD     64
#define BATCH  2
#define SEQ    2048
#define MROWS  (BATCH*SEQ)

#if __has_builtin(__builtin_amdgcn_tensor_load_to_lds) && \
    __has_builtin(__builtin_amdgcn_s_wait_tensorcnt)
#define USE_TDM 1
#else
#define USE_TDM 0
#endif

union FragH {
  v16h v;
  v8h  h[2];
};

__device__ __forceinline__ v8f wmma_f16(v16h a, v16h b, v8f c) {
  // D = A(16x32 f16) x B(32x16 f16) + C(16x16 f32)
  return __builtin_amdgcn_wmma_f32_16x16x32_f16(
      /*neg_a=*/false, a, /*neg_b=*/false, b,
      /*c_mod=*/(short)0, c, /*reuse_a=*/false, /*reuse_b=*/false);
}

__device__ __forceinline__ v8h cvt_v8h(const float4 a, const float4 b) {
  v8h r;
  r[0]=(_Float16)a.x; r[1]=(_Float16)a.y; r[2]=(_Float16)a.z; r[3]=(_Float16)a.w;
  r[4]=(_Float16)b.x; r[5]=(_Float16)b.y; r[6]=(_Float16)b.z; r[7]=(_Float16)b.w;
  return r;
}

#if USE_TDM
// ---------------------------------------------------------------------------
// TDM: DMA a 2-D tile (tile_dw_x DWORDs wide, tile_rows rows, row stride
// row_stride_dw DWORDs) from global to LDS (rows packed contiguously).
// Descriptor bitfields per CDNA5 ISA sec 8.3/8.4 (D# groups 0..3).
// This toolchain exposes the 6-argument builtin form.
// ---------------------------------------------------------------------------
__device__ __forceinline__ void tdm_load_2d(
    unsigned lds_byte_addr, const void* gptr,
    unsigned tile_dw_x, unsigned tile_rows,
    unsigned tensor_dw_x, unsigned tensor_rows,
    unsigned row_stride_dw)
{
  const unsigned long long ga = (unsigned long long)(uintptr_t)gptr;
  u32x4 g0;
  g0[0] = 1u;                                    // count=1, user-mode, no gather
  g0[1] = lds_byte_addr;                         // lds_addr [63:32]
  g0[2] = (unsigned)(ga & 0xffffffffu);          // global_addr [95:64]
  g0[3] = (unsigned)((ga >> 32) & 0x1ffffffu)    // global_addr [120:96]
        | (2u << 30);                            // type=2 ("image") [127:126]
  i32x8 g1;
  g1[0] = (int)(2u << 16);                       // data_size=2 -> 4-byte units
  g1[1] = (int)((tensor_dw_x & 0xffffu) << 16);  // tensor_dim0[15:0] @ [63:48]
  g1[2] = (int)(((tensor_dw_x >> 16) & 0xffffu)  // tensor_dim0[31:16]
        | ((tensor_rows & 0xffffu) << 16));      // tensor_dim1[15:0]
  g1[3] = (int)(((tensor_rows >> 16) & 0xffffu)  // tensor_dim1[31:16]
        | ((tile_dw_x & 0xffffu) << 16));        // tile_dim0 @ [127:112]
  g1[4] = (int)(tile_rows & 0xffffu);            // tile_dim1 ; tile_dim2=0
  g1[5] = (int)row_stride_dw;                    // tensor_dim0_stride[31:0]
  g1[6] = 0;                                     // stride[47:32]=0, dim1_stride=0
  g1[7] = 0;
  i32x4 gz4 = {};                                // groups 2/3 unused (2-D tile)
  i32x8 gz8 = {};
  __builtin_amdgcn_tensor_load_to_lds(g0, g1, gz4, gz4, gz8, 0);
}
#endif

// ---------------------------------------------------------------------------
// fp32 -> f16 conversion (weights), 4 elements / thread
// ---------------------------------------------------------------------------
__global__ __launch_bounds__(256) void cvt_f32_to_f16(
    const float* __restrict__ src, _Float16* __restrict__ dst, int n4) {
  int i = blockIdx.x * blockDim.x + threadIdx.x;
  if (i < n4) {
    const float4 f = ((const float4*)src)[i];
    v4h o;
    o[0]=(_Float16)f.x; o[1]=(_Float16)f.y; o[2]=(_Float16)f.z; o[3]=(_Float16)f.w;
    ((v4h*)dst)[i] = o;
  }
}

// ---------------------------------------------------------------------------
// GEMM: Out[m,n] = A[m,:] . W[n,:] + bias[n]     (y = x @ W^T + b, "NT" form)
// M=4096, N=K=1024. Block tile 128x128, BK=32, 256 threads = 8 waves (2x4).
// Each wave computes 64x32 via 4x2 WMMA 16x16x32 tiles.
// Weight tile is staged LDS-side by the Tensor Data Mover (wave 0), while all
// waves stage/convert the A tile with vector loads in parallel.
// ---------------------------------------------------------------------------
template<bool A_F32, bool OUT_F32, bool V_T>
__global__ __launch_bounds__(256) void gemm_nt_wmma(
    const void* __restrict__ Ap, const _Float16* __restrict__ Bw,
    const float* __restrict__ bias, void* __restrict__ Out)
{
  constexpr int BM = 128, BN = 128, BK = 32;
  __shared__ _Float16 As[BM * BK];
  __shared__ _Float16 Bs[BN * BK];

  const int tid  = threadIdx.x;
  const int wave = tid >> 5;
  const int lane = tid & 31;
  const int l16  = lane & 15;
  const int lh   = lane >> 4;
  const int wm   = wave >> 2;   // 0..1 : 64 rows each
  const int wn   = wave & 3;    // 0..3 : 32 cols each
  const int bm   = blockIdx.x * BM;
  const int bn   = blockIdx.y * BN;

  const int srow = tid >> 1;           // 0..127
  const int scol = (tid & 1) << 4;     // 0 or 16

  v8f acc[4][2] = {};

  for (int kt = 0; kt < HIDDEN; kt += BK) {
    __syncthreads();
    // ---- stage B tile (weights): Tensor Data Mover when available ----
#if USE_TDM
    if (wave == 0) {
      tdm_load_2d((unsigned)(uintptr_t)&Bs[0],
                  (const void*)(Bw + (size_t)bn * HIDDEN + kt),
                  /*tile_dw_x=*/BK / 2, /*tile_rows=*/BN,
                  /*tensor_dw_x=*/HIDDEN / 2, /*tensor_rows=*/HIDDEN,
                  /*row_stride_dw=*/HIDDEN / 2);
    }
#else
    {
      const _Float16* bp = Bw + (size_t)(bn + srow) * HIDDEN + kt + scol;
      *(v8h*)&Bs[srow * BK + scol    ] = *(const v8h*)bp;
      *(v8h*)&Bs[srow * BK + scol + 8] = *(const v8h*)(bp + 8);
      if (kt + BK < HIDDEN) __builtin_prefetch((const void*)(bp + BK), 0, 1);
    }
#endif
    // ---- stage A tile (128x32 f16 in LDS), overlaps the TDM transfer ----
    if (A_F32) {
      const float* ap = (const float*)Ap + (size_t)(bm + srow) * HIDDEN + kt + scol;
      const float4 f0 = ((const float4*)ap)[0];
      const float4 f1 = ((const float4*)ap)[1];
      const float4 f2 = ((const float4*)ap)[2];
      const float4 f3 = ((const float4*)ap)[3];
      *(v8h*)&As[srow * BK + scol    ] = cvt_v8h(f0, f1);
      *(v8h*)&As[srow * BK + scol + 8] = cvt_v8h(f2, f3);
      if (kt + BK < HIDDEN) __builtin_prefetch((const void*)(ap + BK), 0, 1);
    } else {
      const _Float16* ap = (const _Float16*)Ap + (size_t)(bm + srow) * HIDDEN + kt + scol;
      *(v8h*)&As[srow * BK + scol    ] = *(const v8h*)ap;
      *(v8h*)&As[srow * BK + scol + 8] = *(const v8h*)(ap + 8);
      if (kt + BK < HIDDEN) __builtin_prefetch((const void*)(ap + BK), 0, 1);
    }
#if USE_TDM
    if (wave == 0) __builtin_amdgcn_s_wait_tensorcnt(0);
#endif
    __syncthreads();

    // ---- per-wave fragments ----
    FragH a[4], b[2];
#pragma unroll
    for (int i = 0; i < 4; i++) {
      // A layout: lane m=l%16; K 0..7 / 16..23 (lanes<16), K 8..15 / 24..31 (lanes>=16)
      const _Float16* p = &As[(wm * 64 + i * 16 + l16) * BK + lh * 8];
      a[i].h[0] = *(const v8h*)p;
      a[i].h[1] = *(const v8h*)(p + 16);
    }
#pragma unroll
    for (int j = 0; j < 2; j++) {
      // B layout: lane n=l%16; K 0..7 / 16..23 (lanes<16), K 8..15 / 24..31 (lanes>=16)
      const _Float16* p = &Bs[(wn * 32 + j * 16 + l16) * BK + lh * 8];
      b[j].h[0] = *(const v8h*)p;
      b[j].h[1] = *(const v8h*)(p + 16);
    }
#pragma unroll
    for (int i = 0; i < 4; i++)
#pragma unroll
      for (int j = 0; j < 2; j++)
        acc[i][j] = wmma_f16(a[i].v, b[j].v, acc[i][j]);
  }

  // ---- epilogue: C/D layout lane n=l%16, row = r + 8*(lane/16) ----
#pragma unroll
  for (int j = 0; j < 2; j++) {
    const int col = bn + wn * 32 + j * 16 + l16;
    const float bv = bias[col];
#pragma unroll
    for (int i = 0; i < 4; i++) {
#pragma unroll
      for (int r = 0; r < 8; r++) {
        const int row = bm + wm * 64 + i * 16 + r + 8 * lh;
        const float v = acc[i][j][r] + bv;
        if (OUT_F32) {
          ((float*)Out)[(size_t)row * HIDDEN + col] = v;
        } else if (!V_T) {
          ((_Float16*)Out)[(size_t)row * HIDDEN + col] = (_Float16)v;
        } else {
          // row = b*SEQ+s ; col = h*HD+d  ->  Vt[b][h][d][s]
          const int b_ = row >> 11, s_ = row & (SEQ - 1);
          const int h_ = col >> 6,  d_ = col & (HD - 1);
          ((_Float16*)Out)[((size_t)(b_ * HEADS + h_) * HD + d_) * SEQ + s_] = (_Float16)v;
        }
      }
    }
  }
}

// ---------------------------------------------------------------------------
// Flash attention forward. Block = 128 threads (4 waves), one block per
// (batch, head, 64-query tile). Each wave owns 16 query rows x d=64.
// Q/K: [b,s,h*64+d] f16 row-major.  Vt: [b,h,d,s] f16.  Ctx: [b,s,h*64+d].
// ---------------------------------------------------------------------------
__global__ __launch_bounds__(128) void attn_fwd(
    const _Float16* __restrict__ Q, const _Float16* __restrict__ K,
    const _Float16* __restrict__ Vt, _Float16* __restrict__ Ctx)
{
  __shared__ _Float16 Pl[4][16 * 64];   // per-wave P tile (16 q-rows x 64 keys)
  const int wave = threadIdx.x >> 5;
  const int lane = threadIdx.x & 31;
  const int l16  = lane & 15, lh = lane >> 4;
  const int qb = blockIdx.x * 64;
  const int h  = blockIdx.y;
  const int b  = blockIdx.z;
  _Float16* Pw = Pl[wave];

  // Q fragments (16x64 = two 16x32 A-fragments), kept resident all kernel
  const _Float16* qrow = Q + (size_t)(b * SEQ + qb + wave * 16 + l16) * HIDDEN + h * HD;
  FragH qa[2];
#pragma unroll
  for (int kf = 0; kf < 2; kf++) {
    qa[kf].h[0] = *(const v8h*)(qrow + kf * 32 + lh * 8);
    qa[kf].h[1] = *(const v8h*)(qrow + kf * 32 + lh * 8 + 16);
  }

  v8f accO[4] = {};
  float mi[8], li[8];
#pragma unroll
  for (int r = 0; r < 8; r++) { mi[r] = -1e30f; li[r] = 0.0f; }
  const float scale = 0.125f;   // 1/sqrt(64)

  for (int kb = 0; kb < SEQ; kb += 64) {
    // ---- S = Q . K^T  (4 key sub-tiles of 16) ----
    v8f sc[4] = {};
#pragma unroll
    for (int t = 0; t < 4; t++) {
      const _Float16* krow =
          K + (size_t)(b * SEQ + kb + t * 16 + l16) * HIDDEN + h * HD;
#pragma unroll
      for (int kf = 0; kf < 2; kf++) {
        FragH kbf;   // B operand: lane n=key, K=d contiguous
        kbf.h[0] = *(const v8h*)(krow + kf * 32 + lh * 8);
        kbf.h[1] = *(const v8h*)(krow + kf * 32 + lh * 8 + 16);
        sc[t] = wmma_f16(qa[kf].v, kbf.v, sc[t]);
      }
    }
    // ---- online softmax per row (row r+8*lh lives across 16 lanes) ----
#pragma unroll
    for (int r = 0; r < 8; r++) {
      float rowmax = -1e30f;
#pragma unroll
      for (int t = 0; t < 4; t++) { sc[t][r] *= scale; rowmax = fmaxf(rowmax, sc[t][r]); }
#pragma unroll
      for (int m = 1; m < 16; m <<= 1) rowmax = fmaxf(rowmax, __shfl_xor(rowmax, m, 32));
      const float mnew  = fmaxf(mi[r], rowmax);
      const float alpha = __expf(mi[r] - mnew);
      float psum = 0.0f;
#pragma unroll
      for (int t = 0; t < 4; t++) {
        const float p = __expf(sc[t][r] - mnew);
        psum += p;
        Pw[(r + 8 * lh) * 64 + t * 16 + l16] = (_Float16)p;   // P[m][k] to LDS
      }
#pragma unroll
      for (int m = 1; m < 16; m <<= 1) psum += __shfl_xor(psum, m, 32);
      li[r] = li[r] * alpha + psum;
      mi[r] = mnew;
#pragma unroll
      for (int t = 0; t < 4; t++) accO[t][r] *= alpha;
    }
    __syncthreads();   // order P stores before A-fragment reloads

    // ---- O += P . V  ----
    FragH pa[2];
#pragma unroll
    for (int kf = 0; kf < 2; kf++) {
      const _Float16* p = &Pw[l16 * 64 + kf * 32 + lh * 8];
      pa[kf].h[0] = *(const v8h*)p;
      pa[kf].h[1] = *(const v8h*)(p + 16);
    }
#pragma unroll
    for (int td = 0; td < 4; td++) {
      const _Float16* vrow =
          Vt + ((size_t)(b * HEADS + h) * HD + td * 16 + l16) * SEQ + kb;
#pragma unroll
      for (int kf = 0; kf < 2; kf++) {
        FragH vb;   // B operand: lane n=d, K=key contiguous (Vt is [d][s])
        vb.h[0] = *(const v8h*)(vrow + kf * 32 + lh * 8);
        vb.h[1] = *(const v8h*)(vrow + kf * 32 + lh * 8 + 16);
        accO[td] = wmma_f16(pa[kf].v, vb.v, accO[td]);
      }
    }
  }

  // ---- normalize + store ctx ----
#pragma unroll
  for (int td = 0; td < 4; td++) {
    const int col = h * HD + td * 16 + l16;
#pragma unroll
    for (int r = 0; r < 8; r++) {
      const int s_ = qb + wave * 16 + r + 8 * lh;
      Ctx[(size_t)(b * SEQ + s_) * HIDDEN + col] = (_Float16)(accO[td][r] / li[r]);
    }
  }
}

// ---------------------------------------------------------------------------
extern "C" void kernel_launch(void* const* d_in, const int* in_sizes, int n_in,
                              void* d_out, int out_size, void* d_ws, size_t ws_size,
                              hipStream_t stream) {
  (void)in_sizes; (void)n_in; (void)out_size; (void)ws_size;
  const float* q  = (const float*)d_in[0];
  const float* k  = (const float*)d_in[1];
  const float* v  = (const float*)d_in[2];
  const float* Wq = (const float*)d_in[3];
  const float* bq = (const float*)d_in[4];
  const float* Wk = (const float*)d_in[5];
  const float* bk = (const float*)d_in[6];
  const float* Wv = (const float*)d_in[7];
  const float* bv = (const float*)d_in[8];
  const float* Wo = (const float*)d_in[9];
  const float* bo = (const float*)d_in[10];

  const size_t WELEM = (size_t)HIDDEN * HIDDEN;   // 1M
  const size_t XELEM = (size_t)MROWS * HIDDEN;    // 4M
  _Float16* Wq16 = (_Float16*)d_ws;
  _Float16* Wk16 = Wq16 + WELEM;
  _Float16* Wv16 = Wk16 + WELEM;
  _Float16* Wo16 = Wv16 + WELEM;
  _Float16* Q16  = Wo16 + WELEM;
  _Float16* K16  = Q16  + XELEM;
  _Float16* Vt16 = K16  + XELEM;
  _Float16* C16  = Vt16 + XELEM;     // total ~40 MB

  {
    dim3 g((unsigned)((WELEM / 4 + 255) / 256));
    cvt_f32_to_f16<<<g, 256, 0, stream>>>(Wq, Wq16, (int)(WELEM / 4));
    cvt_f32_to_f16<<<g, 256, 0, stream>>>(Wk, Wk16, (int)(WELEM / 4));
    cvt_f32_to_f16<<<g, 256, 0, stream>>>(Wv, Wv16, (int)(WELEM / 4));
    cvt_f32_to_f16<<<g, 256, 0, stream>>>(Wo, Wo16, (int)(WELEM / 4));
  }

  dim3 gg(MROWS / 128, HIDDEN / 128);
  gemm_nt_wmma<true,  false, false><<<gg, 256, 0, stream>>>(q,   Wq16, bq, Q16);
  gemm_nt_wmma<true,  false, false><<<gg, 256, 0, stream>>>(k,   Wk16, bk, K16);
  gemm_nt_wmma<true,  false, true ><<<gg, 256, 0, stream>>>(v,   Wv16, bv, Vt16);

  attn_fwd<<<dim3(SEQ / 64, HEADS, BATCH), 128, 0, stream>>>(Q16, K16, Vt16, C16);

  gemm_nt_wmma<false, true,  false><<<gg, 256, 0, stream>>>(C16, Wo16, bo, d_out);
}
